// DeconvNet1_4166118277830
// MI455X (gfx1250) — matrix operand order repaired and checked by reading, not compiled, not run on hardware
//
#include <hip/hip_runtime.h>
#include <hip/hip_bf16.h>
#include <math.h>

// ---------------------------------------------------------------------------
// GAT autoencoder for MI455X (gfx1250, wave32).
//   enc1 GEMM (WMMA f16) -> f1/f2 -> sparse attn+ELU -> enc2 GEMM (WMMA) -> Z
//   concat(Z, slice_emb) -> dec1 GEMM (WMMA) -> f1/f2 -> sparse attn+ELU
//   -> dec2 GEMM (WMMA) -> recon
// Adjacency is compacted to CSR once (ballot-based, wave32) and reused by
// both GAT layers and all heads.
// ---------------------------------------------------------------------------

#define NN   4096
#define GG   1024
#define HH1  512
#define HH2  64
#define NHD  4
#define CHD  128          // HH1 / NHD
#define SED  16
#define KDEC 80           // HH2 + SED
#define KDECP 96          // padded to multiple of 32
#define CAP  128          // max neighbors kept per row (E[deg]~17)

typedef __attribute__((ext_vector_type(16))) _Float16 v16h;
typedef __attribute__((ext_vector_type(8)))  _Float16 v8h;
typedef __attribute__((ext_vector_type(8)))  float    v8f;

static __device__ inline v8f wmma_f16(v16h a, v16h b, v8f c) {
  return __builtin_amdgcn_wmma_f32_16x16x32_f16(
      /*neg_a=*/false, a, /*neg_b=*/false, b,
      /*c_mod=*/(short)0, c, /*reuse_a=*/false, /*reuse_b=*/false);
}

// ---------------- CSR build: one wave32 per row, ballot compaction ----------
__global__ void build_csr_kernel(const float* __restrict__ adj,
                                 int* __restrict__ idx, int* __restrict__ deg) {
  const int row  = blockIdx.x;
  const int lane = threadIdx.x;           // blockDim.x == 32
  const float* a = adj + (size_t)row * NN;
  int count = 0;
  for (int j0 = 0; j0 < NN; j0 += 32) {
    const float v = a[j0 + lane];
    const unsigned m = (unsigned)__ballot(v != 0.0f);   // wave32: low 32 bits
    if (v != 0.0f) {
      const int pos = count + __popc(m & ((1u << lane) - 1u));
      if (pos < CAP) idx[(size_t)row * CAP + pos] = j0 + lane;
    }
    count += __popc(m);
  }
  if (lane == 0) deg[row] = (count < CAP) ? count : CAP;
}

// ---------------- f32 -> f16 elementwise ------------------------------------
__global__ void cvt_f16_kernel(const float* __restrict__ x,
                               _Float16* __restrict__ y, int n) {
  const int i = blockIdx.x * blockDim.x + threadIdx.x;
  if (i < n) y[i] = (_Float16)x[i];
}

// W[NH][K][CHD] -> BT[(h*CHD+c)][Kpad] (f16, zero-padded K)
__global__ void pack_head_bt_kernel(const float* __restrict__ W,
                                    _Float16* __restrict__ BT,
                                    int K, int Kpad) {
  const int i = blockIdx.x * blockDim.x + threadIdx.x;
  const int total = HH1 * Kpad;
  if (i >= total) return;
  const int n = i / Kpad, k = i % Kpad;
  const int h = n >> 7, c = n & (CHD - 1);
  const float v = (k < K) ? W[((size_t)h * K + k) * CHD + c] : 0.0f;
  BT[i] = (_Float16)v;
}

// W[K][N] -> BT[N][Kpad] (f16)
__global__ void pack_bt_kernel(const float* __restrict__ W,
                               _Float16* __restrict__ BT,
                               int K, int N, int Kpad) {
  const int i = blockIdx.x * blockDim.x + threadIdx.x;
  if (i >= N * Kpad) return;
  const int n = i / Kpad, k = i % Kpad;
  const float v = (k < K) ? W[(size_t)k * N + n] : 0.0f;
  BT[i] = (_Float16)v;
}

// ---------------- WMMA GEMM: C[M,N] = A[M,K]f16 @ BT[N,K]f16^T + bias -------
// block = 256 threads = 8 waves; wave tile 32x32 (2x2 wmma), block tile 128x64
__global__ void wmma_gemm_kernel(const _Float16* __restrict__ A,
                                 const _Float16* __restrict__ BT,
                                 const float* __restrict__ bias,
                                 float* __restrict__ C,
                                 int M, int N, int K) {
  const int lane = threadIdx.x & 31;
  const int wid  = threadIdx.x >> 5;
  const int m0 = blockIdx.y * 128 + (wid >> 1) * 32;
  const int n0 = blockIdx.x * 64  + (wid & 1) * 32;
  const int hi  = lane >> 4;     // half-select
  const int l15 = lane & 15;

  const _Float16* Ar0 = A  + (size_t)(m0 + l15)      * K;
  const _Float16* Ar1 = A  + (size_t)(m0 + 16 + l15) * K;
  const _Float16* Br0 = BT + (size_t)(n0 + l15)      * K;
  const _Float16* Br1 = BT + (size_t)(n0 + 16 + l15) * K;

  v8f acc[2][2] = {};
  for (int k0 = 0; k0 < K; k0 += 32) {
    // A frag: halves 0..7 = A[m][k0+hi*8 ..], 8..15 = A[m][k0+16+hi*8 ..]
    v8h a0l = *(const v8h*)(Ar0 + k0 + hi * 8);
    v8h a0h = *(const v8h*)(Ar0 + k0 + 16 + hi * 8);
    v8h a1l = *(const v8h*)(Ar1 + k0 + hi * 8);
    v8h a1h = *(const v8h*)(Ar1 + k0 + 16 + hi * 8);
    // B frag: halves 0..7 = BT[n][k0+hi*8 ..], 8..15 = BT[n][k0+16+hi*8 ..]
    v8h b0l = *(const v8h*)(Br0 + k0 + hi * 8);
    v8h b0h = *(const v8h*)(Br0 + k0 + 16 + hi * 8);
    v8h b1l = *(const v8h*)(Br1 + k0 + hi * 8);
    v8h b1h = *(const v8h*)(Br1 + k0 + 16 + hi * 8);
    const v16h a0 = __builtin_shufflevector(a0l, a0h, 0,1,2,3,4,5,6,7,8,9,10,11,12,13,14,15);
    const v16h a1 = __builtin_shufflevector(a1l, a1h, 0,1,2,3,4,5,6,7,8,9,10,11,12,13,14,15);
    const v16h b0 = __builtin_shufflevector(b0l, b0h, 0,1,2,3,4,5,6,7,8,9,10,11,12,13,14,15);
    const v16h b1 = __builtin_shufflevector(b1l, b1h, 0,1,2,3,4,5,6,7,8,9,10,11,12,13,14,15);
    acc[0][0] = wmma_f16(a0, b0, acc[0][0]);
    acc[0][1] = wmma_f16(a0, b1, acc[0][1]);
    acc[1][0] = wmma_f16(a1, b0, acc[1][0]);
    acc[1][1] = wmma_f16(a1, b1, acc[1][1]);
  }
  // C/D layout: lane(l15=n, hi), VGPR v -> row m0+hi*8+v, col n0+l15
  #pragma unroll
  for (int tm = 0; tm < 2; ++tm)
    #pragma unroll
    for (int tn = 0; tn < 2; ++tn) {
      const int col = n0 + tn * 16 + l15;
      const float bv = bias ? bias[col] : 0.0f;
      #pragma unroll
      for (int v = 0; v < 8; ++v) {
        const int row = m0 + tm * 16 + hi * 8 + v;
        C[(size_t)row * N + col] = acc[tm][tn][v] + bv;
      }
    }
}

// ---------------- per-head attention scalars: f1 = h.v0, f2 = h.v1 ----------
__global__ void gat_fs_kernel(const float* __restrict__ Hpre,
                              const float* __restrict__ v0,
                              const float* __restrict__ v1,
                              float* __restrict__ f1, float* __restrict__ f2) {
  const int w    = (int)((blockIdx.x * blockDim.x + threadIdx.x) >> 5);
  const int lane = threadIdx.x & 31;
  const int i = w >> 2, h = w & 3;
  if (i >= NN) return;
  const float* hr  = Hpre + (size_t)i * HH1 + h * CHD;
  const float* v0h = v0 + h * CHD;
  const float* v1h = v1 + h * CHD;
  float s1 = 0.f, s2 = 0.f;
  for (int c = lane; c < CHD; c += 32) {
    const float x = hr[c];
    s1 += x * v0h[c];
    s2 += x * v1h[c];
  }
  for (int m = 16; m; m >>= 1) {
    s1 += __shfl_xor(s1, m, 32);
    s2 += __shfl_xor(s2, m, 32);
  }
  if (lane == 0) { f1[h * NN + i] = s1; f2[h * NN + i] = s2; }
}

// ---------------- sparse masked softmax + aggregate + ELU + cvt f16 ---------
// block = 128 threads = 4 waves; wave h computes head-h softmax row; then all
// threads aggregate the 512 output channels (col>>7 selects head).
__global__ void gat_attn_kernel(const float* __restrict__ Hpre,
                                const float* __restrict__ f1,
                                const float* __restrict__ f2,
                                const int* __restrict__ csr,
                                const int* __restrict__ deg,
                                _Float16* __restrict__ out16) {
  __shared__ float pbuf[NHD][CAP];
  __shared__ int   nbr[CAP];
  const int i    = blockIdx.x;
  const int tid  = threadIdx.x;
  const int wid  = tid >> 5;            // head
  const int lane = tid & 31;
  const int d = deg[i];
  for (int t = tid; t < d; t += 128) nbr[t] = csr[(size_t)i * CAP + t];
  __syncthreads();

  // pass 1: s = sigmoid(f1_i + f2_j) - 0.5 (TEMP=1), masked where s == 0
  const float fi = f1[wid * NN + i];
  float mx = -1e9f;                     // matches where(mask, s, -1e9)
  for (int t = lane; t < d; t += 32) {
    const float lg = fi + f2[wid * NN + nbr[t]];
    const float u  = 1.0f / (1.0f + __expf(-lg)) - 0.5f;
    const float s  = (u != 0.0f) ? u : -3.0e38f;   // sentinel for masked
    pbuf[wid][t] = s;
    if (u != 0.0f) mx = fmaxf(mx, s);
  }
  for (int m = 16; m; m >>= 1) mx = fmaxf(mx, __shfl_xor(mx, m, 32));

  float den = 0.f;
  for (int t = lane; t < d; t += 32) {
    const float s = pbuf[wid][t];
    const float e = (s > -1e30f) ? __expf(s - mx) : 0.0f;
    pbuf[wid][t] = e;
    den += e;
  }
  for (int m = 16; m; m >>= 1) den += __shfl_xor(den, m, 32);
  const float inv = 1.0f / fmaxf(den, 1e-30f);
  for (int t = lane; t < d; t += 32) pbuf[wid][t] *= inv;
  __syncthreads();

  // pass 2: out[i, col] = ELU( sum_j p[h][j] * Hpre[nbr[j], col] )
  for (int col = tid; col < HH1; col += 128) {
    const int h = col >> 7;
    float acc = 0.f;
    for (int t = 0; t < d; ++t)
      acc += pbuf[h][t] * Hpre[(size_t)nbr[t] * HH1 + col];
    const float y = (acc > 0.f) ? acc : (expm1f(acc));
    out16[(size_t)i * HH1 + col] = (_Float16)y;
  }
}

// ---------------- decoder input: concat(Z, emb[slice]) padded to 96, f16 ----
__global__ void build_dec_in_kernel(const float* __restrict__ Z,
                                    const int* __restrict__ slice,
                                    const float* __restrict__ emb,
                                    _Float16* __restrict__ Hd) {
  const int idx = blockIdx.x * blockDim.x + threadIdx.x;
  if (idx >= NN * KDECP) return;
  const int i = idx / KDECP, c = idx % KDECP;
  float v = 0.0f;
  if (c < HH2)       v = Z[(size_t)i * HH2 + c];
  else if (c < KDEC) v = emb[slice[i] * SED + (c - HH2)];
  Hd[idx] = (_Float16)v;
}

// ---------------------------------------------------------------------------
extern "C" void kernel_launch(void* const* d_in, const int* in_sizes, int n_in,
                              void* d_out, int out_size, void* d_ws, size_t ws_size,
                              hipStream_t stream) {
  (void)in_sizes; (void)n_in; (void)out_size; (void)ws_size;
  const float* adj     = (const float*)d_in[0];
  const float* nodeF   = (const float*)d_in[1];
  const int*   slice   = (const int*)  d_in[4];
  const float* enc1_W  = (const float*)d_in[6];
  const float* enc1_b  = (const float*)d_in[7];   // [4,128] flat == [512]
  const float* enc1_v0 = (const float*)d_in[8];
  const float* enc1_v1 = (const float*)d_in[9];
  const float* enc2_W  = (const float*)d_in[10];
  const float* enc2_b  = (const float*)d_in[11];
  const float* dec1_W  = (const float*)d_in[12];
  const float* dec1_b  = (const float*)d_in[13];
  const float* dec1_v0 = (const float*)d_in[14];
  const float* dec1_v1 = (const float*)d_in[15];
  const float* dec2_W  = (const float*)d_in[16];
  const float* dec2_b  = (const float*)d_in[17];
  const float* emb     = (const float*)d_in[18];

  float* recon = (float*)d_out;                      // [4096,1024]
  float* Zout  = (float*)d_out + (size_t)NN * GG;    // [4096,64]

  // workspace layout (all offsets multiples of 256 B)
  char* ws = (char*)d_ws;
  _Float16* X16     = (_Float16*)(ws + 0);           //  8 MB
  _Float16* enc1BT  = (_Float16*)(ws +  8388608);    //  1 MB   [512][1024]
  _Float16* enc2BT  = (_Float16*)(ws +  9437184);    // 64 KB   [64][512]
  _Float16* dec1BT  = (_Float16*)(ws +  9502720);    // 96 KB   [512][96]
  _Float16* dec2BT  = (_Float16*)(ws +  9601024);    //  1 MB   [1024][512]
  int*      csr     = (int*)     (ws + 10649600);    //  2 MB   [4096][128]
  int*      deg     = (int*)     (ws + 12746752);    // 16 KB
  float*    Hpre    = (float*)   (ws + 12763136);    //  8 MB   [4096][512]
  float*    f1      = (float*)   (ws + 21151744);    // 64 KB   [4][4096]
  float*    f2      = (float*)   (ws + 21217280);    // 64 KB
  _Float16* Hattn16 = (_Float16*)(ws + 21282816);    //  4 MB   [4096][512]
  _Float16* Hd16    = (_Float16*)(ws + 25477120);    // 768 KB  [4096][96]
  float*    Hdpre   = (float*)   (ws + 26263552);    //  8 MB   [4096][512]
  float*    f1d     = (float*)   (ws + 34652160);    // 64 KB
  float*    f2d     = (float*)   (ws + 34717696);    // 64 KB
  _Float16* Hdattn16= (_Float16*)(ws + 34783232);    //  4 MB   (end 38,977,536)

  // --- preprocessing (CSR + f16 packing) ---
  build_csr_kernel<<<NN, 32, 0, stream>>>(adj, csr, deg);
  {
    const int n = NN * GG;
    cvt_f16_kernel<<<(n + 255) / 256, 256, 0, stream>>>(nodeF, X16, n);
  }
  pack_head_bt_kernel<<<(HH1 * GG + 255) / 256, 256, 0, stream>>>(enc1_W, enc1BT, GG, GG);
  pack_head_bt_kernel<<<(HH1 * KDECP + 255) / 256, 256, 0, stream>>>(dec1_W, dec1BT, KDEC, KDECP);
  pack_bt_kernel<<<(HH2 * HH1 + 255) / 256, 256, 0, stream>>>(enc2_W, enc2BT, HH1, HH2, HH1);
  pack_bt_kernel<<<(GG * HH1 + 255) / 256, 256, 0, stream>>>(dec2_W, dec2BT, HH1, GG, HH1);

  // --- encoder ---
  {
    dim3 g(HH1 / 64, NN / 128);
    wmma_gemm_kernel<<<g, 256, 0, stream>>>(X16, enc1BT, enc1_b, Hpre, NN, HH1, GG);
  }
  gat_fs_kernel<<<(NHD * NN) / 8, 256, 0, stream>>>(Hpre, enc1_v0, enc1_v1, f1, f2);
  gat_attn_kernel<<<NN, 128, 0, stream>>>(Hpre, f1, f2, csr, deg, Hattn16);
  {
    dim3 g(HH2 / 64, NN / 128);   // Z -> d_out tail
    wmma_gemm_kernel<<<g, 256, 0, stream>>>(Hattn16, enc2BT, enc2_b, Zout, NN, HH2, HH1);
  }

  // --- decoder ---
  build_dec_in_kernel<<<(NN * KDECP + 255) / 256, 256, 0, stream>>>(Zout, slice, emb, Hd16);
  {
    dim3 g(HH1 / 64, NN / 128);
    wmma_gemm_kernel<<<g, 256, 0, stream>>>(Hd16, dec1BT, dec1_b, Hdpre, NN, HH1, KDECP);
  }
  gat_fs_kernel<<<(NHD * NN) / 8, 256, 0, stream>>>(Hdpre, dec1_v0, dec1_v1, f1d, f2d);
  gat_attn_kernel<<<NN, 128, 0, stream>>>(Hdpre, f1d, f2d, csr, deg, Hdattn16);
  {
    dim3 g(GG / 64, NN / 128);    // recon -> d_out head
    wmma_gemm_kernel<<<g, 256, 0, stream>>>(Hdattn16, dec2BT, dec2_b, recon, NN, GG, HH1);
  }
}
